// GATLayer_77730318123229
// MI455X (gfx1250) — compile-verified, matrix operand order checked by reading, not run on hardware
//
#include <hip/hip_runtime.h>
#include <hip/hip_bf16.h>

typedef __attribute__((ext_vector_type(16))) _Float16 v16h;
typedef __attribute__((ext_vector_type(8)))  float    v8f;

#define GAT_ALPHA   0.2f
#define GAT_THRESH  0.5f
#define GAT_NEGINF  -9.0e15f
#define GAT_B 32
#define GAT_N 1024
#define GAT_C 64

#if defined(__gfx1250__) && __has_builtin(__builtin_amdgcn_tensor_load_to_lds) && \
    __has_builtin(__builtin_amdgcn_s_wait_tensorcnt)
#define GAT_HAS_TDM 1
typedef __attribute__((ext_vector_type(4))) unsigned tdm_v4u;
typedef __attribute__((ext_vector_type(8))) int      tdm_v8i;
typedef __attribute__((ext_vector_type(4))) int      tdm_v4i;
#else
#define GAT_HAS_TDM 0
#endif

// ---------------------------------------------------------------------------
// Kernel 1: h = X @ W   (X: (B*N, C) f32, W: (C, C) f32) -> h f16 (B*N, C)
// One workgroup (4 waves) per 16-row tile; wave w owns 16-col tile w.
// K=64 handled as two 16x16x32 WMMAs.
// ---------------------------------------------------------------------------
__global__ __launch_bounds__(128) void gat_h_kernel(
    const float* __restrict__ X, const float* __restrict__ W,
    _Float16* __restrict__ hOut)
{
    const int rowBase = blockIdx.x * 16;
    const int wv      = threadIdx.x >> 5;
    const int lane    = threadIdx.x & 31;
    const int m       = lane & 15;
    const int hs      = lane >> 4;      // half-select (lanes 0-15 vs 16-31)
    const int cBase   = wv * 16;

    v8f acc = {};
#pragma unroll
    for (int k0 = 0; k0 < GAT_C; k0 += 32) {
        v16h aF, bF;
#pragma unroll
        for (int v = 0; v < 8; ++v) {
            const int k = k0 + ((v & 4) ? 16 : 0) + 8 * hs + 2 * (v & 3);
            const float2 xa = *(const float2*)(X + (size_t)(rowBase + m) * GAT_C + k);
            aF[2 * v]     = (_Float16)xa.x;
            aF[2 * v + 1] = (_Float16)xa.y;
            bF[2 * v]     = (_Float16)W[(size_t)k       * GAT_C + cBase + m];
            bF[2 * v + 1] = (_Float16)W[(size_t)(k + 1) * GAT_C + cBase + m];
        }
        acc = __builtin_amdgcn_wmma_f32_16x16x32_f16(
            false, aF, false, bF, (short)0, acc, false, false);
    }
#pragma unroll
    for (int r = 0; r < 8; ++r) {
        const int row = rowBase + r + 8 * hs;
        hOut[(size_t)row * GAT_C + cBase + m] = (_Float16)acc[r];
    }
}

// ---------------------------------------------------------------------------
// Kernel 2: s_src[row] = h[row,:]·a[0:C], s_dst[row] = h[row,:]·a[C:2C]
// ---------------------------------------------------------------------------
__global__ __launch_bounds__(256) void gat_s_kernel(
    const _Float16* __restrict__ h, const float* __restrict__ a,
    float* __restrict__ sSrc, float* __restrict__ sDst)
{
    const int row = blockIdx.x * blockDim.x + threadIdx.x;   // 0..B*N-1
    const _Float16* hr = h + (size_t)row * GAT_C;
    float ss = 0.f, sd = 0.f;
#pragma unroll 8
    for (int c = 0; c < GAT_C; ++c) {
        const float hv = (float)hr[c];
        ss += hv * a[c];
        sd += hv * a[GAT_C + c];
    }
    sSrc[row] = ss;
    sDst[row] = sd;
}

// ---------------------------------------------------------------------------
// Kernel 3: fused masked softmax + attention@h + ELU.
// grid = (N/16, B), block = 128 (4 waves). Two-pass softmax; pass 2 is a
// double-buffered LDS pipeline over 64-wide j-blocks: the h tile is DMA'd
// into LDS by the Tensor Data Mover (one tensor_load_to_lds per iteration,
// waited with s_wait_tensorcnt), while all threads compute the exp() p-tile.
// ---------------------------------------------------------------------------
__global__ __launch_bounds__(128) void gat_attn_kernel(
    const float* __restrict__ adj,
    const _Float16* __restrict__ h,
    const float* __restrict__ sSrc,
    const float* __restrict__ sDst,
    float* __restrict__ out)
{
    __shared__ float    redM[16][8];
    __shared__ float    redL[16][8];
    __shared__ float    mRow[16];
    __shared__ float    lInv[16];
    __shared__ _Float16 pT[2][16][64];      // exp(e - m) tiles (A operand)
    __shared__ _Float16 hT[2][64][GAT_C];   // h tiles        (B operand)

    const int iBase = blockIdx.x * 16;
    const int b     = blockIdx.y;
    const int tid   = threadIdx.x;

    // ---- Pass 1: per-row running max / sum over all j (float4 loads) ----
    {
        const int row = tid >> 3;         // 0..15
        const int sub = tid & 7;          // 8 threads per row
        const int i   = iBase + row;
        const float si = sSrc[b * GAT_N + i];
        const float* adjRow = adj + (size_t)i * GAT_N;
        const float* sdB    = sDst + (size_t)b * GAT_N;
        float m = -3.0e38f, l = 0.f;
        for (int j0 = sub * 4; j0 < GAT_N; j0 += 32) {
            const float4 av = *(const float4*)(adjRow + j0);
            const float4 sv = *(const float4*)(sdB + j0);
            const float ax[4] = {av.x, av.y, av.z, av.w};
            const float sx[4] = {sv.x, sv.y, sv.z, sv.w};
            float e[4];
#pragma unroll
            for (int jj = 0; jj < 4; ++jj) {
                const float x  = si + sx[jj];
                const float lk = x > 0.f ? x : GAT_ALPHA * x;
                e[jj] = ax[jj] > GAT_THRESH ? lk : GAT_NEGINF;
            }
            const float em = fmaxf(fmaxf(e[0], e[1]), fmaxf(e[2], e[3]));
            const float nm = fmaxf(m, em);
            l = l * __expf(m - nm) + __expf(e[0] - nm) + __expf(e[1] - nm)
                                   + __expf(e[2] - nm) + __expf(e[3] - nm);
            m = nm;
        }
        redM[row][sub] = m;
        redL[row][sub] = l;
    }
    __syncthreads();
    if (tid < 16) {
        float M = -3.0e38f;
#pragma unroll
        for (int s = 0; s < 8; ++s) M = fmaxf(M, redM[tid][s]);
        float L = 0.f;
#pragma unroll
        for (int s = 0; s < 8; ++s) L += redL[tid][s] * __expf(redM[tid][s] - M);
        mRow[tid] = M;
        lInv[tid] = 1.0f / L;
    }
    __syncthreads();

    // ---- Pass 2: acc += exp(e - m) @ h via WMMA, 64-wide j pipeline -----
    const int lane  = tid & 31;
    const int wv    = tid >> 5;
    const int mIdx  = lane & 15;
    const int hs    = lane >> 4;
    const int cBase = wv * 16;
    const _Float16* hB = h + (size_t)b * GAT_N * GAT_C;

    // p tile: 16x64, 8 values per thread, one 16B LDS store; prefetch ahead
    auto stageP = [&](int buf, int jbn) {
        const int row = tid >> 3;
        const int j0  = (tid & 7) * 8;
        const int i   = iBase + row;
        const float si = sSrc[b * GAT_N + i];
        const float M  = mRow[row];
        alignas(16) _Float16 pv[8];
#pragma unroll
        for (int q = 0; q < 2; ++q) {
            const float4 av = *(const float4*)(adj + (size_t)i * GAT_N + jbn + j0 + 4 * q);
            const float4 sv = *(const float4*)(sDst + (size_t)b * GAT_N + jbn + j0 + 4 * q);
            const float ax[4] = {av.x, av.y, av.z, av.w};
            const float sx[4] = {sv.x, sv.y, sv.z, sv.w};
#pragma unroll
            for (int jj = 0; jj < 4; ++jj) {
                const float x  = si + sx[jj];
                const float lk = x > 0.f ? x : GAT_ALPHA * x;
                const float e  = ax[jj] > GAT_THRESH ? lk : GAT_NEGINF;
                pv[4 * q + jj] = (_Float16)__expf(e - M);
            }
        }
        *(uint4*)&pT[buf][row][j0] = *(const uint4*)pv;
        if (jbn + 64 < GAT_N)
            __builtin_prefetch(adj + (size_t)i * GAT_N + jbn + 64 + j0, 0, 0);
    };

#if GAT_HAS_TDM
    // h tile via Tensor Data Mover: one DMA of a 64x64 f16 2D tile into LDS.
    auto loadH = [&](int buf, int jbn) {
        if (wv == 0) {
            const unsigned lds =
                (unsigned)(unsigned long long)(const void*)&hT[buf][0][0];
            const unsigned long long ga =
                (unsigned long long)(const void*)(hB + (size_t)jbn * GAT_C);
            // D# group 0: count=1 | lds_addr | global_addr[56:0] | type=2
            tdm_v4u g0 = { 1u, lds, (unsigned)ga,
                           (unsigned)((ga >> 32) & 0x01FFFFFFull) | (2u << 30) };
            // D# group 1: data_size=2B; tensor_dim0=64; tensor_dim1=1024;
            //             tile_dim0=64; tile_dim1=64; dim0_stride=64
            tdm_v8i g1 = { (int)0x00010000,
                           (int)(64u << 16),
                           (int)(1024u << 16),
                           (int)(64u << 16),
                           64,
                           64,
                           0, 0 };
            tdm_v4i gz = { 0, 0, 0, 0 };              // 2D tensor: groups 2/3 unused
            tdm_v8i z8 = { 0, 0, 0, 0, 0, 0, 0, 0 };  // extra operand (clang-23 form)
            __builtin_amdgcn_tensor_load_to_lds(g0, g1, gz, gz, z8, 0);
        }
    };
#else
    // fallback: cooperative copy, 512 uint4 -> 4 per thread
    auto loadH = [&](int buf, int jbn) {
        const uint4* src = (const uint4*)(hB + (size_t)jbn * GAT_C);
        uint4* dst = (uint4*)&hT[buf][0][0];
#pragma unroll
        for (int q = 0; q < 4; ++q) dst[tid + 128 * q] = src[tid + 128 * q];
    };
#endif

    v8f acc = {};
    stageP(0, 0);
    loadH(0, 0);
    int p = 0;
    for (int jb = 0; jb < GAT_N; jb += 64) {
#if GAT_HAS_TDM
        if (wv == 0) __builtin_amdgcn_s_wait_tensorcnt(0);  // DMA landed
#endif
        __syncthreads();                         // staged buffer p is ready
        if (jb + 64 < GAT_N) {                   // overlap next stage
            stageP(p ^ 1, jb + 64);
            loadH(p ^ 1, jb + 64);
        }
#pragma unroll
        for (int kt = 0; kt < 2; ++kt) {
            v16h aF, bF;
#pragma unroll
            for (int v = 0; v < 8; ++v) {
                const int k = 32 * kt + ((v & 4) ? 16 : 0) + 8 * hs + 2 * (v & 3);
                aF[2 * v]     = pT[p][mIdx][k];
                aF[2 * v + 1] = pT[p][mIdx][k + 1];
                bF[2 * v]     = hT[p][k][cBase + mIdx];
                bF[2 * v + 1] = hT[p][k + 1][cBase + mIdx];
            }
            acc = __builtin_amdgcn_wmma_f32_16x16x32_f16(
                false, aF, false, bF, (short)0, acc, false, false);
        }
        p ^= 1;
    }

    // ---- Epilogue: normalize by 1/l, ELU, store -------------------------
#pragma unroll
    for (int r = 0; r < 8; ++r) {
        const int row = r + 8 * hs;
        float val = acc[r] * lInv[row];
        val = val > 0.f ? val : (__expf(val) - 1.0f);
        out[((size_t)(b * GAT_N + iBase + row)) * GAT_C + cBase + mIdx] = val;
    }
}

// ---------------------------------------------------------------------------
extern "C" void kernel_launch(void* const* d_in, const int* in_sizes, int n_in,
                              void* d_out, int out_size, void* d_ws, size_t ws_size,
                              hipStream_t stream)
{
    const float* X   = (const float*)d_in[0];   // (32,1024,64)
    const float* adj = (const float*)d_in[1];   // (1024,1024)
    const float* W   = (const float*)d_in[2];   // (64,64)
    const float* a   = (const float*)d_in[3];   // (128,1)
    float* out = (float*)d_out;

    char* ws = (char*)d_ws;
    _Float16* h  = (_Float16*)ws;                                   // 4 MB
    float* sSrc  = (float*)(ws + (size_t)4 * 1024 * 1024);          // 128 KB
    float* sDst  = (float*)(ws + (size_t)4 * 1024 * 1024 + 131072); // 128 KB

    const int M = GAT_B * GAT_N;                 // 32768 rows

    gat_h_kernel<<<M / 16, 128, 0, stream>>>(X, W, h);
    gat_s_kernel<<<M / 256, 256, 0, stream>>>(h, a, sSrc, sDst);
    dim3 g3(GAT_N / 16, GAT_B);
    gat_attn_kernel<<<g3, 128, 0, stream>>>(adj, h, sSrc, sDst, out);
}
